// ResidualLSTM_72516227826020
// MI455X (gfx1250) — matrix operand order repaired and checked by reading, not compiled, not run on hardware
//
#include <hip/hip_runtime.h>
#include <hip/hip_bf16.h>
#include <math.h>

// Problem dims (fixed by the reference)
#define SS 512
#define BB 128
#define II 1024
#define HH 1024
#define GG 4096              // 4*H
#define M1 (SS * BB)         // 65536 rows for the batched input GEMM

typedef __attribute__((ext_vector_type(16))) _Float16 v16h;
typedef __attribute__((ext_vector_type(8)))  _Float16 v8h;
typedef __attribute__((ext_vector_type(8)))  float    v8f;

// ---------------------------------------------------------------------------
// Helpers
// ---------------------------------------------------------------------------
__device__ __forceinline__ float sigmoidf_fast(float x) {
    return 1.0f / (1.0f + __expf(-x));
}
// tanh via the hardware transcendental: 1 - 2/(1+e^{2x}).
// Saturates correctly: x>>0 -> e^{2x}=inf -> 1;  x<<0 -> e^{2x}=0 -> -1.
__device__ __forceinline__ float tanhf_fast(float x) {
    return 1.0f - 2.0f / (1.0f + __expf(2.0f * x));
}

// Load a 16-bit A-fragment (16x32, M x K) for wave32 WMMA.
// ISA layout: lanes 0-15 hold row M=lane, K = {k..k+7, k+16..k+23};
//             lanes 16-31 hold row M=lane-16, K = {k+8..k+15, k+24..k+31}.
__device__ __forceinline__ v16h load_a_frag(const _Float16* __restrict__ arow,
                                            int k, int koff) {
    v8h lo = *(const v8h*)(arow + k + koff);        // 16B aligned
    v8h hi = *(const v8h*)(arow + k + koff + 16);   // 16B aligned
    v16h a;
#pragma unroll
    for (int i = 0; i < 8; ++i) { a[i] = lo[i]; a[8 + i] = hi[i]; }
    return a;
}

// ---------------------------------------------------------------------------
// One-time data movement / precision conversion
// ---------------------------------------------------------------------------
__global__ void cvt_f32_to_f16(const float* __restrict__ src,
                               _Float16* __restrict__ dst, size_t n) {
    size_t i = (size_t)blockIdx.x * blockDim.x + threadIdx.x;
    if (i < n) dst[i] = (_Float16)src[i];
}

// w: [G][K] row-major f32  ->  wT: [K][G] row-major f16  (B-matrix friendly)
__global__ void transpose_w_f16(const float* __restrict__ w,
                                _Float16* __restrict__ wT,
                                int gdim, int kdim) {
    size_t i = (size_t)blockIdx.x * blockDim.x + threadIdx.x;
    size_t n = (size_t)gdim * kdim;
    if (i < n) {
        int g = (int)(i / kdim);
        int k = (int)(i % kdim);
        wT[(size_t)k * gdim + g] = (_Float16)w[i];
    }
}

__global__ void init_state(_Float16* __restrict__ h0, float* __restrict__ c) {
    size_t i = (size_t)blockIdx.x * blockDim.x + threadIdx.x;
    if (i < (size_t)BB * HH) { h0[i] = (_Float16)0.0f; c[i] = 0.0f; }
}

// ---------------------------------------------------------------------------
// Phase 1: gates_x = x @ w_ih^T + b_ih + b_hh     ([M1 x I] x [I x G])
// grid = (G/128, M1/128), block = 256 (8 waves). Each wave: one 16-row M-tile,
// EIGHT 16-col N-tiles. Explicitly software-pipelined: B(q+1) is loaded
// before WMMA(q) issues, and the next k-chunk's A fragment is loaded before
// the WMMA burst, so loads stay in flight behind the matrix ops.
// ---------------------------------------------------------------------------
__global__ void __launch_bounds__(256)
lstm_gemm_x(const _Float16* __restrict__ x16,   // [M1][I]
            const _Float16* __restrict__ wT,    // [I][G]
            const float* __restrict__ b_ih,     // [G]
            const float* __restrict__ b_hh,     // [G]
            float* __restrict__ gx)             // [M1][G]
{
    const int lane  = threadIdx.x & 31;
    const int wave  = threadIdx.x >> 5;
    const int nbase = blockIdx.x * 128;
    const int mbase = (blockIdx.y * 8 + wave) * 16;

    const int row  = mbase + (lane & 15);
    const int koff = (lane >> 4) * 8;

    v8f acc[8];
#pragma unroll
    for (int q = 0; q < 8; ++q) {
        const int col = nbase + q * 16 + (lane & 15);
        const float bias = b_ih[col] + b_hh[col];
#pragma unroll
        for (int v = 0; v < 8; ++v) acc[q][v] = bias;
    }

    const _Float16* arow = x16 + (size_t)row * II;
    v16h a = load_a_frag(arow, 0, koff);                // prologue A
    for (int k = 0; k < II; k += 32) {
        // Prefetch streamed A ~16 chunks ahead into near caches
        // (stays inside the contiguous workspace region).
        __builtin_prefetch(arow + k + 512, 0, 3);

        // Pipelined A for next chunk (dead copy on final iteration).
        v16h an = a;
        if (k + 32 < II) an = load_a_frag(arow, k + 32, koff);

        // B fragments: lane holds col N = lane&15, K = {k+koff..+7, k+koff+16..+23}.
        const _Float16* brow = wT + (size_t)(k + koff) * GG + nbase + (lane & 15);
        v16h bq;                                        // prologue B(0)
#pragma unroll
        for (int i = 0; i < 8; ++i) { bq[i] = brow[(size_t)i * GG]; bq[8 + i] = brow[(size_t)(16 + i) * GG]; }
#pragma unroll
        for (int q = 0; q < 8; ++q) {
            v16h bn = bq;
            if (q < 7) {
#pragma unroll
                for (int i = 0; i < 8; ++i) { bn[i] = brow[(size_t)i * GG + (q + 1) * 16]; bn[8 + i] = brow[(size_t)(16 + i) * GG + (q + 1) * 16]; }
            }  // B(q+1)
            acc[q] = __builtin_amdgcn_wmma_f32_16x16x32_f16(
                false, a, false, bq, (short)0, acc[q], false, false);
            bq = bn;
        }
        a = an;
    }

    // C layout: lane l, vgpr v -> row mbase + v + 8*(l>>4), col nbase+q*16+(l&15)
    const int r0 = mbase + (lane >> 4) * 8;
#pragma unroll
    for (int q = 0; q < 8; ++q) {
        const int col = nbase + q * 16 + (lane & 15);
#pragma unroll
        for (int v = 0; v < 8; ++v)
            gx[(size_t)(r0 + v) * GG + col] = acc[q][v];
    }
}

// ---------------------------------------------------------------------------
// Phase 2: one recurrent step.
// gates = gx[t] + h_in @ w_hh^T ; fused sigmoid/tanh cell update.
// grid = (H/16, 2), block = 128 (4 waves). 128 WGs spread the latency-bound
// step over more WGPs. Each wave computes the SAME 16 hidden cols for all 4
// gate quadrants (N offsets j, H+j, 2H+j, 3H+j) so the combine is lane-local.
// Same software pipelining as phase 1 (this is the sequential critical path).
// ---------------------------------------------------------------------------
__global__ void __launch_bounds__(128)
lstm_step(const _Float16* __restrict__ h_in,  // [B][H] f16 (prev step)
          _Float16* __restrict__ h_out,       // [B][H] f16 (this step)
          float* __restrict__ c,              // [B][H] f32 cell state
          const _Float16* __restrict__ whhT,  // [H][G] f16
          const float* __restrict__ gx_t,     // [B][G] f32 (precomputed)
          float* __restrict__ out_t)          // [B][H] f32 -> d_out[t]
{
    const int lane  = threadIdx.x & 31;
    const int wave  = threadIdx.x >> 5;                 // 0..3
    const int jbase = blockIdx.x * 16;                  // hidden-column tile
    const int mbase = (blockIdx.y * 4 + wave) * 16;     // batch-row tile

    const int row  = mbase + (lane & 15);
    const int koff = (lane >> 4) * 8;

    // Initialize accumulators with gx[t] (folds the "+ gates_x" add into C).
    const int r0 = mbase + (lane >> 4) * 8;
    const int cl = lane & 15;
    v8f acc[4];
#pragma unroll
    for (int q = 0; q < 4; ++q) {
        const int col = q * HH + jbase + cl;
#pragma unroll
        for (int v = 0; v < 8; ++v)
            acc[q][v] = gx_t[(size_t)(r0 + v) * GG + col];
    }

    const _Float16* arow = h_in + (size_t)row * HH;
    v16h a = load_a_frag(arow, 0, koff);                // prologue A
    for (int k = 0; k < HH; k += 32) {
        v16h an = a;
        if (k + 32 < HH) an = load_a_frag(arow, k + 32, koff);

        const _Float16* brow = whhT + (size_t)(k + koff) * GG + jbase + cl;
        v16h bq;                                        // quadrant 0
#pragma unroll
        for (int i = 0; i < 8; ++i) { bq[i] = brow[(size_t)i * GG]; bq[8 + i] = brow[(size_t)(16 + i) * GG]; }
#pragma unroll
        for (int q = 0; q < 4; ++q) {
            v16h bn = bq;
            if (q < 3) {
#pragma unroll
                for (int i = 0; i < 8; ++i) { bn[i] = brow[(size_t)i * GG + (q + 1) * HH]; bn[8 + i] = brow[(size_t)(16 + i) * GG + (q + 1) * HH]; }
            }  // next quadrant
            acc[q] = __builtin_amdgcn_wmma_f32_16x16x32_f16(
                false, a, false, bq, (short)0, acc[q], false, false);
            bq = bn;
        }
        a = an;
    }

    // Fused gate epilogue: i,f,g,o -> c,h update (fast transcendentals).
    const int j = jbase + cl;
#pragma unroll
    for (int v = 0; v < 8; ++v) {
        const int    brow_ = r0 + v;
        const size_t idx   = (size_t)brow_ * HH + j;
        const float ig = sigmoidf_fast(acc[0][v]);
        const float fg = sigmoidf_fast(acc[1][v]);
        const float gg = tanhf_fast(acc[2][v]);
        const float og = sigmoidf_fast(acc[3][v]);
        const float cn = fg * c[idx] + ig * gg;
        const float hn = og * tanhf_fast(cn);
        c[idx]     = cn;
        h_out[idx] = (_Float16)hn;
        out_t[idx] = hn;
    }
}

// Copy final hidden / cell state into the hN / cN output slots.
__global__ void finalize(const float* __restrict__ h_last,
                         const float* __restrict__ c,
                         float* __restrict__ hN, float* __restrict__ cN) {
    size_t i = (size_t)blockIdx.x * blockDim.x + threadIdx.x;
    if (i < (size_t)BB * HH) { hN[i] = h_last[i]; cN[i] = c[i]; }
}

// ---------------------------------------------------------------------------
// Launcher
// ---------------------------------------------------------------------------
extern "C" void kernel_launch(void* const* d_in, const int* in_sizes, int n_in,
                              void* d_out, int out_size, void* d_ws, size_t ws_size,
                              hipStream_t stream) {
    (void)in_sizes; (void)n_in; (void)out_size; (void)ws_size;

    const float* x    = (const float*)d_in[0];  // [S,B,I]
    const float* w_ih = (const float*)d_in[1];  // [4H,I]
    const float* w_hh = (const float*)d_in[2];  // [4H,H]
    const float* b_ih = (const float*)d_in[3];  // [4H]
    const float* b_hh = (const float*)d_in[4];  // [4H]
    float* out = (float*)d_out;                 // [S,B,H] ++ hN[B,H] ++ cN[B,H]

    // Workspace carve-up (all regions fully initialized below every call).
    char* p = (char*)d_ws;
    float*    gx   = (float*)p;     p += (size_t)M1 * GG * sizeof(float);     // 1 GiB
    _Float16* x16  = (_Float16*)p;  p += (size_t)M1 * II * sizeof(_Float16);  // 128 MiB
    _Float16* wihT = (_Float16*)p;  p += (size_t)II * GG * sizeof(_Float16);  // 8 MiB
    _Float16* whhT = (_Float16*)p;  p += (size_t)HH * GG * sizeof(_Float16);  // 8 MiB
    _Float16* hbuf0 = (_Float16*)p; p += (size_t)BB * HH * sizeof(_Float16);
    _Float16* hbuf1 = (_Float16*)p; p += (size_t)BB * HH * sizeof(_Float16);
    float*    cbuf  = (float*)p;    p += (size_t)BB * HH * sizeof(float);

    // --- one-time conversions / transposes (parallel, bandwidth-bound) ---
    {
        const size_t n = (size_t)M1 * II;
        cvt_f32_to_f16<<<(unsigned)((n + 255) / 256), 256, 0, stream>>>(x, x16, n);
    }
    transpose_w_f16<<<(unsigned)(((size_t)GG * II + 255) / 256), 256, 0, stream>>>(
        w_ih, wihT, GG, II);
    transpose_w_f16<<<(unsigned)(((size_t)GG * HH + 255) / 256), 256, 0, stream>>>(
        w_hh, whhT, GG, HH);
    init_state<<<(unsigned)(((size_t)BB * HH + 255) / 256), 256, 0, stream>>>(
        hbuf0, cbuf);

    // --- phase 1: batched input GEMM across all timesteps (WMMA) ---
    lstm_gemm_x<<<dim3(GG / 128, M1 / 128), 256, 0, stream>>>(
        x16, wihT, b_ih, b_hh, gx);

    // --- phase 2: sequential recurrence, one small WMMA kernel per step ---
    for (int t = 0; t < SS; ++t) {
        const _Float16* hi = (t & 1) ? hbuf1 : hbuf0;
        _Float16*       ho = (t & 1) ? hbuf0 : hbuf1;
        lstm_step<<<dim3(HH / 16, 2), 128, 0, stream>>>(
            hi, ho, cbuf, whhT,
            gx  + (size_t)t * BB * GG,
            out + (size_t)t * BB * HH);
    }

    // --- outputs: hN = output[S-1], cN = cell state ---
    finalize<<<(unsigned)(((size_t)BB * HH + 255) / 256), 256, 0, stream>>>(
        out + (size_t)(SS - 1) * BB * HH, cbuf,
        out + (size_t)SS * BB * HH,
        out + (size_t)SS * BB * HH + (size_t)BB * HH);
}
